// CausalSelfAttention_21912923144942
// MI455X (gfx1250) — matrix operand order repaired and checked by reading, not compiled, not run on hardware
//
#include <hip/hip_runtime.h>

#define B_   2
#define T_   2048
#define C_   1024
#define NH_  16
#define NKV_ 4
#define HD_  64

typedef __attribute__((ext_vector_type(16))) __bf16 v16bf;
typedef __attribute__((ext_vector_type(8)))  __bf16 v8bf;
typedef __attribute__((ext_vector_type(8)))  float  v8f;

#define SHUF16(lo, hi) __builtin_shufflevector((lo), (hi), 0,1,2,3,4,5,6,7,8,9,10,11,12,13,14,15)

static __device__ __forceinline__ __bf16 f2bf(float f) {
  unsigned u = __builtin_bit_cast(unsigned, f);
  unsigned r = u + 0x7FFFu + ((u >> 16) & 1u);   // round-to-nearest-even
  unsigned short h = (unsigned short)(r >> 16);
  return __builtin_bit_cast(__bf16, h);
}

// A fragment (16x32 bf16), row-major source. Lane holds row (lane&15).
// elem e in [0,8): K = half*8 + e ; e in [8,16): K = 16 + half*8 + (e-8)
static __device__ __forceinline__ v16bf load_frag_a(const __bf16* row, int half) {
  v8bf lo = *(const v8bf*)(row + half * 8);
  v8bf hi = *(const v8bf*)(row + 16 + half * 8);
  return SHUF16(lo, hi);
}
// B fragment (32x16 bf16), source is transposed (per-column contiguous over K).
// Lane holds column (lane&15); elem e in [0,8): K = half*8 + e ; e in [8,16): K = 16 + half*8 + (e-8)
static __device__ __forceinline__ v16bf load_frag_b(const __bf16* col, int half) {
  v8bf lo = *(const v8bf*)(col + half * 8);
  v8bf hi = *(const v8bf*)(col + 16 + half * 8);
  return SHUF16(lo, hi);
}

static __device__ __forceinline__ float rowmax16(float v) {
#pragma unroll
  for (int m = 1; m < 16; m <<= 1) v = fmaxf(v, __shfl_xor(v, m, 32));
  return v;
}
static __device__ __forceinline__ float rowsum16(float v) {
#pragma unroll
  for (int m = 1; m < 16; m <<= 1) v += __shfl_xor(v, m, 32);
  return v;
}

// ---------------------------------------------------------------------------
// elementwise fp32 -> bf16
__global__ void convert_f32_to_bf16(const float* __restrict__ src, __bf16* __restrict__ dst, int n) {
  int i = blockIdx.x * blockDim.x + threadIdx.x;
  if (i < n) dst[i] = f2bf(src[i]);
}

// src: R x Ccol fp32 (row-major) -> dst: Ccol x R bf16 (row-major)
__global__ void transpose_f32_to_bf16(const float* __restrict__ src, __bf16* __restrict__ dst,
                                      int R, int Ccol) {
  int i = blockIdx.x * blockDim.x + threadIdx.x;
  if (i >= R * Ccol) return;
  int r = i / Ccol, c = i - r * Ccol;
  dst[(size_t)c * R + r] = f2bf(src[i]);
}

// ---------------------------------------------------------------------------
// C(MxN fp32) = A(MxK bf16, row-major) @ B, with B given transposed: BT(NxK bf16).
// One wave per 32x64 C tile: 2 A-fragments x 4 B-fragments -> 8 WMMA accumulators.
// (A reused 4x, B reused 2x => ~21 flops/byte vs 8 for a 16x16 tile.)
__global__ void wmma_gemm_bf16(const __bf16* __restrict__ A, const __bf16* __restrict__ BT,
                               float* __restrict__ C, int M, int N, int K) {
  int n0 = blockIdx.x * 64;
  int m0 = blockIdx.y * 32;
  int lane = threadIdx.x;
  int half = lane >> 4, r = lane & 15;
  const __bf16* ap0 = A + (size_t)(m0 + r) * K;
  const __bf16* ap1 = A + (size_t)(m0 + 16 + r) * K;
  const __bf16* bp  = BT + (size_t)(n0 + r) * K;
  v8f acc[2][4] = {};
  for (int k0 = 0; k0 < K; k0 += 32) {
    v16bf a0 = load_frag_a(ap0 + k0, half);
    v16bf a1 = load_frag_a(ap1 + k0, half);
#pragma unroll
    for (int j = 0; j < 4; ++j) {
      v16bf b = load_frag_b(bp + (size_t)j * 16 * K + k0, half);
      acc[0][j] = __builtin_amdgcn_wmma_f32_16x16x32_bf16(false, a0, false, b, (short)0,
                                                          acc[0][j], false, false);
      acc[1][j] = __builtin_amdgcn_wmma_f32_16x16x32_bf16(false, a1, false, b, (short)0,
                                                          acc[1][j], false, false);
    }
  }
#pragma unroll
  for (int mi = 0; mi < 2; ++mi) {
#pragma unroll
    for (int j = 0; j < 4; ++j) {
      float* cp = C + (size_t)(m0 + mi * 16 + half * 8) * N + n0 + j * 16 + r;
#pragma unroll
      for (int i = 0; i < 8; ++i) cp[(size_t)i * N] = acc[mi][j][i];
    }
  }
}

// ---------------------------------------------------------------------------
// gate / v-merge / RoPE / RMS-norm; reshuffle into attention-friendly bf16 layouts.
// grid.x = B*T, block = 32 threads (task 0..15: q heads, 16..19: k heads, 20..23: v heads)
__global__ void postprocess_kernel(const float* __restrict__ x, const float* __restrict__ ve,
                                   const float* __restrict__ cosb, const float* __restrict__ sinb,
                                   const float* __restrict__ Wg,
                                   const float* __restrict__ qf, const float* __restrict__ kf,
                                   const float* __restrict__ vf,
                                   __bf16* __restrict__ qb, __bf16* __restrict__ kbuf,
                                   __bf16* __restrict__ vbT) {
  int bt = blockIdx.x;
  int b = bt / T_, t = bt - b * T_;
  int task = threadIdx.x;
  if (task >= NH_ + 2 * NKV_) return;

  if (task < NH_ + NKV_) {
    // q or k head: RoPE then RMS-norm * 1.2
    const float* src;
    __bf16* dst;
    if (task < NH_) {
      int h = task;
      src = qf + (size_t)bt * (NH_ * HD_) + h * HD_;
      dst = qb + (((size_t)(b * NH_ + h) * T_) + t) * HD_;
    } else {
      int h = task - NH_;
      src = kf + (size_t)bt * (NKV_ * HD_) + h * HD_;
      dst = kbuf + (((size_t)(b * NKV_ + h) * T_) + t) * HD_;
    }
    float v[HD_];
#pragma unroll
    for (int d = 0; d < HD_; ++d) v[d] = src[d];
    const float* cp = cosb + (size_t)t * (HD_ / 2);
    const float* sp = sinb + (size_t)t * (HD_ / 2);
#pragma unroll
    for (int d = 0; d < HD_ / 2; ++d) {
      float c = cp[d], s = sp[d];
      float x1 = v[d], x2 = v[d + HD_ / 2];
      v[d] = x1 * c + x2 * s;
      v[d + HD_ / 2] = -x1 * s + x2 * c;
    }
    float ss = 0.f;
#pragma unroll
    for (int d = 0; d < HD_; ++d) ss += v[d] * v[d];
    float sc = rsqrtf(ss * (1.0f / HD_) + 1e-6f) * 1.2f;
#pragma unroll
    for (int d = 0; d < HD_; ++d) dst[d] = f2bf(v[d] * sc);
  } else {
    // v head: v = v + 3*sigmoid(x[:12] @ Wg[:,h]) * ve ; store transposed [d][t]
    int h = task - NH_ - NKV_;
    const float* xr = x + (size_t)bt * C_;
    float g = 0.f;
#pragma unroll
    for (int j = 0; j < 12; ++j) g += xr[j] * Wg[j * NKV_ + h];
    g = 3.0f / (1.0f + __expf(-g));
    const float* vsrc  = vf + (size_t)bt * (NKV_ * HD_) + h * HD_;
    const float* vesrc = ve + (size_t)bt * (NKV_ * HD_) + h * HD_;
    __bf16* dst = vbT + ((size_t)(b * NKV_ + h) * HD_) * T_;
#pragma unroll
    for (int d = 0; d < HD_; ++d) dst[(size_t)d * T_ + t] = f2bf(vsrc[d] + g * vesrc[d]);
  }
}

// ---------------------------------------------------------------------------
// Flash-style sliding-window causal attention.
// grid = (T/16, NH, B); block = 32 (one wave handles a 16-row q tile of one head).
__global__ void attn_kernel(const __bf16* __restrict__ qb, const __bf16* __restrict__ kbuf,
                            const __bf16* __restrict__ vbT, __bf16* __restrict__ yb,
                            const int* __restrict__ wptr) {
  __shared__ __align__(16) __bf16 Plds[16][32];
  const int window = *wptr;
  const int t0 = blockIdx.x * 16;
  const int h = blockIdx.y, b = blockIdx.z;
  const int kh = h / (NH_ / NKV_);
  const int lane = threadIdx.x;
  const int half = lane >> 4, r = lane & 15;

  // Q fragments for both 32-dim k-steps (dims 0..31 and 32..63)
  const __bf16* qp = qb + (((size_t)(b * NH_ + h) * T_) + (t0 + r)) * HD_;
  v16bf qa0 = load_frag_a(qp, half);
  v16bf qa1 = load_frag_a(qp + 32, half);

  float mrow[8], lrow[8];
  v8f o0 = {}, o1 = {}, o2 = {}, o3 = {};
#pragma unroll
  for (int i = 0; i < 8; ++i) { mrow[i] = 0.0f; lrow[i] = 0.0f; }  // scores bounded; 0-floor is safe

  int jlo = t0 - window; if (jlo < 0) jlo = 0; jlo &= ~31;
  const int jhi = t0 + 15;
  const __bf16* kbase = kbuf + ((size_t)(b * NKV_ + kh) * T_) * HD_;
  const __bf16* vbase = vbT + ((size_t)(b * NKV_ + kh) * HD_) * T_;

  for (int j0 = jlo; j0 <= jhi; j0 += 32) {
    float sc[2][8];
#pragma unroll
    for (int kt = 0; kt < 2; ++kt) {
      int jb = j0 + kt * 16;
      int key = jb + r; if (key >= T_) key = T_ - 1;     // clamp load; masked below
      const __bf16* kp = kbase + (size_t)key * HD_;
      v16bf kb0 = load_frag_b(kp, half);                 // dims 0..31
      v16bf kb1 = load_frag_b(kp + 32, half);            // dims 32..63
      v8f s = {};
      s = __builtin_amdgcn_wmma_f32_16x16x32_bf16(false, qa0, false, kb0, (short)0, s, false, false);
      s = __builtin_amdgcn_wmma_f32_16x16x32_bf16(false, qa1, false, kb1, (short)0, s, false, false);
      int kcol = jb + r;
#pragma unroll
      for (int i = 0; i < 8; ++i) {
        int qrow = t0 + half * 8 + i;
        float val = s[i] * 0.125f;                       // 1/sqrt(64)
        bool ok = (kcol <= qrow) && (qrow - kcol <= window) && (kcol < T_);
        sc[kt][i] = ok ? val : -1.0e9f;
      }
    }
    // online softmax (row = half*8+i, striped over the 16 lanes of this half)
    float p0a[8], p1a[8];
#pragma unroll
    for (int i = 0; i < 8; ++i) {
      float rm = rowmax16(fmaxf(sc[0][i], sc[1][i]));
      float newm = fmaxf(mrow[i], rm);
      float corr = __expf(mrow[i] - newm);
      float p0 = __expf(sc[0][i] - newm);
      float p1 = __expf(sc[1][i] - newm);
      lrow[i] = lrow[i] * corr + rowsum16(p0 + p1);
      mrow[i] = newm;
      p0a[i] = p0; p1a[i] = p1;
      o0[i] = o0[i] * corr; o1[i] = o1[i] * corr;
      o2[i] = o2[i] * corr; o3[i] = o3[i] * corr;
    }
    __syncthreads();
#pragma unroll
    for (int i = 0; i < 8; ++i) {
      Plds[half * 8 + i][r]      = f2bf(p0a[i]);
      Plds[half * 8 + i][16 + r] = f2bf(p1a[i]);
    }
    __syncthreads();
    // P as A-fragment: lane row = r, K(e<8)=half*8+e, K(e>=8)=16+half*8+(e-8)
    v8bf plo = *(const v8bf*)&Plds[r][half * 8];
    v8bf phi = *(const v8bf*)&Plds[r][16 + half * 8];
    v16bf pa = SHUF16(plo, phi);
    // P(16x32) @ V(32x64): 4 WMMAs over 16-dim output slices; V stored transposed -> contiguous B frags
    v16bf vb0 = load_frag_b(vbase + (size_t)(0 * 16 + r) * T_ + j0, half);
    v16bf vb1 = load_frag_b(vbase + (size_t)(1 * 16 + r) * T_ + j0, half);
    v16bf vb2 = load_frag_b(vbase + (size_t)(2 * 16 + r) * T_ + j0, half);
    v16bf vb3 = load_frag_b(vbase + (size_t)(3 * 16 + r) * T_ + j0, half);
    o0 = __builtin_amdgcn_wmma_f32_16x16x32_bf16(false, pa, false, vb0, (short)0, o0, false, false);
    o1 = __builtin_amdgcn_wmma_f32_16x16x32_bf16(false, pa, false, vb1, (short)0, o1, false, false);
    o2 = __builtin_amdgcn_wmma_f32_16x16x32_bf16(false, pa, false, vb2, (short)0, o2, false, false);
    o3 = __builtin_amdgcn_wmma_f32_16x16x32_bf16(false, pa, false, vb3, (short)0, o3, false, false);
  }

#pragma unroll
  for (int i = 0; i < 8; ++i) {
    float invl = 1.0f / lrow[i];
    size_t row = (size_t)b * T_ + t0 + half * 8 + i;
    __bf16* yp = yb + row * C_ + h * HD_;
    yp[0 * 16 + r] = f2bf(o0[i] * invl);
    yp[1 * 16 + r] = f2bf(o1[i] * invl);
    yp[2 * 16 + r] = f2bf(o2[i] * invl);
    yp[3 * 16 + r] = f2bf(o3[i] * invl);
  }
}

// ---------------------------------------------------------------------------
extern "C" void kernel_launch(void* const* d_in, const int* in_sizes, int n_in,
                              void* d_out, int out_size, void* d_ws, size_t ws_size,
                              hipStream_t stream) {
  (void)in_sizes; (void)n_in; (void)out_size; (void)ws_size;
  const float* x     = (const float*)d_in[0];
  const float* ve    = (const float*)d_in[1];
  const float* cosb  = (const float*)d_in[2];
  const float* sinb  = (const float*)d_in[3];
  const float* Wq    = (const float*)d_in[4];
  const float* Wk    = (const float*)d_in[5];
  const float* Wv    = (const float*)d_in[6];
  const float* Wproj = (const float*)d_in[7];
  const float* Wg    = (const float*)d_in[8];
  const int*   wptr  = (const int*)d_in[9];

  char* ws = (char*)d_ws;
  size_t off = 0;
  auto take = [&](size_t bytes) -> char* {
    char* p = ws + off;
    off += (bytes + 255) & ~(size_t)255;
    return p;
  };
  const size_t BT = (size_t)B_ * T_;
  __bf16* xb   = (__bf16*)take(BT * C_ * 2);
  __bf16* WqT  = (__bf16*)take((size_t)C_ * C_ * 2);
  __bf16* WkT  = (__bf16*)take((size_t)C_ * (NKV_ * HD_) * 2);
  __bf16* WvT  = (__bf16*)take((size_t)C_ * (NKV_ * HD_) * 2);
  __bf16* WpT  = (__bf16*)take((size_t)C_ * C_ * 2);
  float*  qf   = (float*)take(BT * (NH_ * HD_) * 4);
  float*  kf   = (float*)take(BT * (NKV_ * HD_) * 4);
  float*  vf   = (float*)take(BT * (NKV_ * HD_) * 4);
  __bf16* qb   = (__bf16*)take(BT * NH_ * HD_ * 2);
  __bf16* kbuf = (__bf16*)take(BT * NKV_ * HD_ * 2);
  __bf16* vbT  = (__bf16*)take((BT * NKV_ * HD_ + 64) * 2);  // +pad for tail vector loads
  __bf16* yb   = (__bf16*)take(BT * C_ * 2);

  // 1) precision conversion / weight transposes
  {
    int n = (int)(BT * C_);
    convert_f32_to_bf16<<<dim3((n + 255) / 256), dim3(256), 0, stream>>>(x, xb, n);
  }
  transpose_f32_to_bf16<<<dim3((C_ * C_ + 255) / 256), dim3(256), 0, stream>>>(Wq, WqT, C_, C_);
  transpose_f32_to_bf16<<<dim3((C_ * NKV_ * HD_ + 255) / 256), dim3(256), 0, stream>>>(Wk, WkT, C_, NKV_ * HD_);
  transpose_f32_to_bf16<<<dim3((C_ * NKV_ * HD_ + 255) / 256), dim3(256), 0, stream>>>(Wv, WvT, C_, NKV_ * HD_);
  transpose_f32_to_bf16<<<dim3((C_ * C_ + 255) / 256), dim3(256), 0, stream>>>(Wproj, WpT, C_, C_);

  // 2) QKV projections (WMMA bf16 -> fp32); 32x64 tile per wave
  wmma_gemm_bf16<<<dim3((NH_ * HD_) / 64, (unsigned)(BT / 32)), dim3(32), 0, stream>>>(
      xb, WqT, qf, (int)BT, NH_ * HD_, C_);
  wmma_gemm_bf16<<<dim3((NKV_ * HD_) / 64, (unsigned)(BT / 32)), dim3(32), 0, stream>>>(
      xb, WkT, kf, (int)BT, NKV_ * HD_, C_);
  wmma_gemm_bf16<<<dim3((NKV_ * HD_) / 64, (unsigned)(BT / 32)), dim3(32), 0, stream>>>(
      xb, WvT, vf, (int)BT, NKV_ * HD_, C_);

  // 3) gate + RoPE + RMS-norm + relayout
  postprocess_kernel<<<dim3((unsigned)BT), dim3(32), 0, stream>>>(
      x, ve, cosb, sinb, Wg, qf, kf, vf, qb, kbuf, vbT);

  // 4) windowed-causal attention (WMMA)
  attn_kernel<<<dim3(T_ / 16, NH_, B_), dim3(32), 0, stream>>>(qb, kbuf, vbT, yb, wptr);

  // 5) output projection (WMMA) -> fp32 out
  wmma_gemm_bf16<<<dim3(C_ / 64, (unsigned)(BT / 32)), dim3(32), 0, stream>>>(
      yb, WpT, (float*)d_out, (int)BT, C_, C_);
}
